// Conv_45921790329340
// MI455X (gfx1250) — matrix operand order repaired and checked by reading, not compiled, not run on hardware
//
#include <hip/hip_runtime.h>
#include <hip/hip_bf16.h>

typedef __attribute__((ext_vector_type(16))) _Float16 v16h;
typedef __attribute__((ext_vector_type(8)))  _Float16 v8h;
typedef __attribute__((ext_vector_type(2)))  _Float16 v2h;
typedef __attribute__((ext_vector_type(8)))  float    v8f;

#define KS       7
#define PAD      3
#define IMG_H    512
#define IMG_W    512
#define TILE_Y   32                  // output rows per slab (2 wave-tiles)
#define TILE_X   64                  // output cols per workgroup (4 wave-tiles)
#define SLABS    4                   // slabs per workgroup (128-row strip)
#define HALO_Y   (TILE_Y + KS - 1)   // 38 (even -> row pairs tile exactly)
#define HALO_X   (TILE_X + KS - 1)   // 70
#define COL_H    48                  // LDS column height (halves): 96B stride, 16B aligned
#define PAD_ROWS (COL_H - HALO_Y)    // 10 zero rows (k in [38,48) -> A==0 region)
#define NPAIRS   (HALO_Y / 2)        // 19 pair-rows
#define NTHREADS 256

__global__ __launch_bounds__(NTHREADS) void conv7x7_wmma_kernel(
    const float* __restrict__ X,     // (nimg, 512, 512) f32
    const float* __restrict__ Kw,    // (7,7) f32
    float* __restrict__ Out)         // (nimg, 512, 512) f32
{
    // Transposed halo tile: lds[c * COL_H + rr] = f16(in[y0s-3+rr, x0-3+c])
    __shared__ __align__(16) _Float16 lds[HALO_X * COL_H];
    // Pre-laid-out A fragments: lds_a[(dx*32 + lane)*16 + h]
    __shared__ __align__(32) _Float16 lds_a[KS * 32 * 16];
    __shared__ float kw_s[KS * KS];

    const int tid = threadIdx.x;
    const int img = blockIdx.z;
    const int x0  = blockIdx.x * TILE_X;
    const int ys0 = blockIdx.y * (TILE_Y * SLABS);

    const float* __restrict__ Ximg = X + (size_t)img * IMG_H * IMG_W;
    float* __restrict__ Oimg = Out + (size_t)img * IMG_H * IMG_W;

    if (tid < KS * KS) kw_s[tid] = Kw[tid];
    __syncthreads();

    // ---- Build A fragments cooperatively (identical for all waves/WGs) ----
    // A^(dx)[m,k] = w[k-m, dx] for 0<=k-m<7 else 0.
    // A k order used here (contiguous, same as B): K(lane,h) = h + (lane>=16 ? 16 : 0), M = lane&15.
    for (int e = tid; e < KS * 32 * 16; e += NTHREADS) {
        int dx = e >> 9;              // /512
        int l  = (e >> 4) & 31;
        int h  = e & 15;
        int m  = l & 15;
        int k  = h + ((l >> 4) << 4);
        int dy = k - m;
        float w = (dy >= 0 && dy < KS) ? kw_s[dy * KS + dx] : 0.0f;
        lds_a[e] = (_Float16)w;
    }
    // Zero the pad rows (38..47) of every halo column once; never rewritten.
    for (int e = tid; e < HALO_X * PAD_ROWS; e += NTHREADS) {
        int c = e / PAD_ROWS;
        int r = e - c * PAD_ROWS;
        lds[c * COL_H + HALO_Y + r] = (_Float16)0.0f;
    }
    __syncthreads();

    const int lane = tid & 31;
    const int wave = tid >> 5;      // 0..7
    const int hi   = lane >> 4;     // lane group 0/1
    const int ln   = lane & 15;     // M for A, N for B/C/D
    const int trow = wave >> 2;     // 0..1
    const int tcol = wave & 3;      // 0..3

    v16h afrag[KS];
    #pragma unroll
    for (int dx = 0; dx < KS; ++dx)
        afrag[dx] = *(const v16h*)&lds_a[(dx * 32 + lane) * 16];

    const int rrb = trow * 16 + hi * 16;   // 0,16,32 -> 16B aligned in column
    const bool xin = (x0 >= PAD) && (x0 + TILE_X + PAD <= IMG_W);

    for (int s = 0; s < SLABS; ++s) {
        const int y0s = ys0 + s * TILE_Y;
        const bool fast = xin && (y0s >= PAD) && (y0s + TILE_Y + PAD <= IMG_H);

        // ---- Stage halo slab (f32 -> f16x2, transposed, row-paired) ----
        if (fast) {
            const float* __restrict__ bp =
                Ximg + (size_t)(y0s - PAD) * IMG_W + (x0 - PAD);
            for (int e = tid; e < NPAIRS * HALO_X; e += NTHREADS) {
                int pr = e / HALO_X;             // pair-row 0..18
                int c  = e - pr * HALO_X;        // consecutive tid -> coalesced
                int rr = pr * 2;
                float v0 = bp[rr * IMG_W + c];
                float v1 = bp[(rr + 1) * IMG_W + c];
                v2h p2 = { (_Float16)v0, (_Float16)v1 };
                *(v2h*)&lds[c * COL_H + rr] = p2;   // rr even -> 4B aligned
            }
        } else {
            for (int e = tid; e < NPAIRS * HALO_X; e += NTHREADS) {
                int pr  = e / HALO_X;
                int c   = e - pr * HALO_X;
                int rr  = pr * 2;
                int row = y0s - PAD + rr;
                int col = x0 - PAD + c;
                float v0 = 0.0f, v1 = 0.0f;
                bool cok = (unsigned)col < IMG_W;
                if (cok && (unsigned)row < IMG_H)
                    v0 = Ximg[row * IMG_W + col];
                if (cok && (unsigned)(row + 1) < IMG_H)
                    v1 = Ximg[(row + 1) * IMG_W + col];
                v2h p2 = { (_Float16)v0, (_Float16)v1 };
                *(v2h*)&lds[c * COL_H + rr] = p2;
            }
        }
        __syncthreads();

        // ---- Prefetch ALL B fragments first (one clause of 14 ds_load_b128),
        //      then run the 7-WMMA accumulation chain.
        // B layout: K(lane,h) = h + (lane>=16 ? 16 : 0), N = lane&15.
        v8h bfr[2 * KS];
        #pragma unroll
        for (int dx = 0; dx < KS; ++dx) {
            const _Float16* p = &lds[(tcol * 16 + ln + dx) * COL_H + rrb];
            bfr[2 * dx]     = *(const v8h*)(p);       // K = rrb .. rrb+7
            bfr[2 * dx + 1] = *(const v8h*)(p + 8);   // K = rrb+8 .. rrb+15
        }

        v8f acc = {};
        #pragma unroll
        for (int dx = 0; dx < KS; ++dx) {
            v16h b;
            #pragma unroll
            for (int i = 0; i < 8; ++i) {
                b[i]     = bfr[2 * dx][i];
                b[i + 8] = bfr[2 * dx + 1][i];
            }
            acc = __builtin_amdgcn_wmma_f32_16x16x32_f16(
                false, afrag[dx], false, b, (short)0, acc, false, false);
        }

        // ---- Store D: element v -> row M = v + hi*8, col N = ln ----
        const int orow0 = y0s + trow * 16 + hi * 8;
        const int ocol  = x0 + tcol * 16 + ln;
        float* __restrict__ op = Oimg + (size_t)orow0 * IMG_W + ocol;
        #pragma unroll
        for (int v = 0; v < 8; ++v)
            op[v * IMG_W] = acc[v];

        __syncthreads();   // protect lds before next slab's staging
    }
}

extern "C" void kernel_launch(void* const* d_in, const int* in_sizes, int n_in,
                              void* d_out, int out_size, void* d_ws, size_t ws_size,
                              hipStream_t stream) {
    (void)n_in; (void)out_size; (void)d_ws; (void)ws_size;
    const float* X  = (const float*)d_in[0];
    const float* Kw = (const float*)d_in[1];
    float* Out      = (float*)d_out;

    const int nimg = in_sizes[0] / (IMG_H * IMG_W);            // 64
    dim3 grid(IMG_W / TILE_X, IMG_H / (TILE_Y * SLABS), nimg); // (8, 4, 64)
    conv7x7_wmma_kernel<<<grid, NTHREADS, 0, stream>>>(X, Kw, Out);
}
